// MixtureOfExperts_29755533427503
// MI455X (gfx1250) — matrix operand order repaired and checked by reading, not compiled, not run on hardware
//
#include <hip/hip_runtime.h>
#include <hip/hip_bf16.h>
#include <math.h>

// MoE: B=4, T=4096, D=256, O=256, E=16.  N = B*T = 16384 tokens.
// out[n,o] = sum_e softmax_e(x@Wg + bg)[n,e] * gelu((x@We[e] + be[e]))[n,o]
//
// bf16 WMMA (f32 accum); 64-token blocks for 4x weight reuse; x tile staged in
// LDS (padded rows -> conflict-free ds_load_b128); per-expert B fragments held
// in registers and M-tiles retired sequentially; half/expert loops pinned
// rolled (unroll 1) to keep the wave under 256 VGPRs; hardware-transcendental
// GELU epilogue.
//
// ws layout:
//   xb  : bf16 [16384][256]          @ 0         (8,388,608 B)
//   Wt  : bf16 [16][256(O)][256(D)]  @ 8388608   (2,097,152 B)  (We transposed)
//   WgT : bf16 [16(E)][256(D)]       @ 10485760  (8,192 B)      (Wg transposed)

typedef __attribute__((ext_vector_type(16))) __bf16 v16bf;
typedef __attribute__((ext_vector_type(8)))  __bf16 v8bf;
typedef __attribute__((ext_vector_type(8)))  float  v8f;

#define NTOK 16384
#define DDIM 256
#define ODIM 256
#define NEXP 16
#define XS_STRIDE 264   // 256 + 8 bf16 pad: half-wave lanes hit disjoint bank groups

// ---------------- prep kernels ----------------

__global__ void moe_prep_x(const float* __restrict__ x, __bf16* __restrict__ xb) {
    int i = blockIdx.x * blockDim.x + threadIdx.x;
    xb[i] = (__bf16)x[i];
}

__global__ void moe_prep_we(const float* __restrict__ We, __bf16* __restrict__ Wt) {
    int i = blockIdx.x * blockDim.x + threadIdx.x;
    int d = i & 255;
    int o = (i >> 8) & 255;
    int e = i >> 16;
    Wt[i] = (__bf16)We[(((size_t)e << 8) + d) * 256 + o];   // Wt[e][o][d] = We[e][d][o]
}

__global__ void moe_prep_wg(const float* __restrict__ Wg, __bf16* __restrict__ WgT) {
    int i = blockIdx.x * blockDim.x + threadIdx.x;
    int d = i & 255;
    int e = i >> 8;
    WgT[i] = (__bf16)Wg[d * 16 + e];                        // WgT[e][d] = Wg[d][e]
}

// ---------------- fragment loaders (ISA 16-bit WMMA layouts, wave32) ----------------

// A (16x32): lane row M = lane&15, g = lane>>4; v[0..7]=K(8g..8g+7), v[8..15]=K(16+8g..23+8g)
__device__ __forceinline__ v16bf load_frag_a(const __bf16* rowbase, int k, int g) {
    v8bf lo = *(const v8bf*)(rowbase + k + 8 * g);
    v8bf hi = *(const v8bf*)(rowbase + k + 16 + 8 * g);
    return __builtin_shufflevector(lo, hi, 0, 1, 2, 3, 4, 5, 6, 7,
                                           8, 9, 10, 11, 12, 13, 14, 15);
}

// B (32x16): lane col N = lane&15, g = lane>>4; v[0..7]=K(8g..8g+7), v[8..15]=K(16+8g..23+8g)
__device__ __forceinline__ v16bf load_frag_b(const __bf16* colbase, int k, int g) {
    v8bf lo = *(const v8bf*)(colbase + k + 8 * g);
    v8bf hi = *(const v8bf*)(colbase + k + 16 + 8 * g);
    return __builtin_shufflevector(lo, hi, 0, 1, 2, 3, 4, 5, 6, 7,
                                           8, 9, 10, 11, 12, 13, 14, 15);
}

// tanh-form GELU on the hardware transcendental pipe (v_exp_f32 + v_rcp_f32):
// 0.5x(1+tanh(c(x+0.044715x^3))) = x / (1 + exp(-2*c*(x+0.044715x^3)))
__device__ __forceinline__ float gelu_fast(float x) {
    float t = x * x;
    float inner = 0.7978845608028654f * x * fmaf(0.044715f, t, 1.0f);
    float e = __expf(-2.0f * inner);
    return x * __builtin_amdgcn_rcpf(1.0f + e);
}

// ---------------- main fused kernel ----------------
// grid = 256 blocks (64 tokens each), block = 256 threads = 8 waves.
// wave w owns O columns [16w,16w+16) then [16w+128,16w+144) for all 4 token M-tiles.

__global__ void __launch_bounds__(256)
moe_main(const __bf16* __restrict__ xb,
         const __bf16* __restrict__ Wt,
         const __bf16* __restrict__ WgT,
         const float*  __restrict__ be,
         const float*  __restrict__ bg,
         float*        __restrict__ out) {
    __shared__ __bf16 xs[64 * XS_STRIDE];        // 33.8 KB staged x tile
    __shared__ float  gateLds[4 * NEXP * 16];    // [mtile][expert][token]

    const int lane = threadIdx.x & 31;
    const int wave = threadIdx.x >> 5;
    const int ln   = lane & 15;
    const int g    = lane >> 4;
    const int tokBase = blockIdx.x * 64;

    // ---- stage x tile (64 x 256 bf16) into LDS, 8 chunks of 16B per thread ----
    #pragma unroll
    for (int c = 0; c < 8; ++c) {
        int chunk = threadIdx.x + c * 256;       // 0..2047
        int row = chunk >> 5;
        int col = (chunk & 31) * 8;
        *(v8bf*)(&xs[row * XS_STRIDE + col]) =
            *(const v8bf*)(xb + (size_t)(tokBase + row) * DDIM + col);
    }
    __syncthreads();

    // ---- gate logits^T per M-tile: waves 0..3, A = WgT (expert rows), B = x cols (LDS) ----
    if (wave < 4) {
        const __bf16* wgrow = WgT + (size_t)ln * DDIM;
        const __bf16* xscol = &xs[(wave * 16 + ln) * XS_STRIDE];
        v8f c;
        #pragma unroll
        for (int r = 0; r < 8; ++r) c[r] = 0.0f;
        #pragma unroll 4                         // limit load pipelining depth here
        for (int ks = 0; ks < 8; ++ks) {
            v16bf a = load_frag_a(wgrow, ks * 32, g);
            v16bf b = load_frag_b(xscol, ks * 32, g);
            c = __builtin_amdgcn_wmma_f32_16x16x32_bf16(false, a, false, b,
                                                        (short)0, c, false, false);
        }
        #pragma unroll
        for (int r = 0; r < 8; ++r) {
            int e = 8 * g + r;                   // C: VGPR r -> M(expert)=r+8g, N(token)=ln
            gateLds[wave * 256 + e * 16 + ln] = c[r] + bg[e];
        }
    }
    __syncthreads();

    // ---- softmax over 16 experts, one token per lane (64 tokens) ----
    if (threadIdx.x < 64) {
        int mt = threadIdx.x >> 4;
        int tt = threadIdx.x & 15;
        float* gl = &gateLds[mt * 256 + tt];
        float mx = gl[0];
        #pragma unroll
        for (int e = 1; e < NEXP; ++e) mx = fmaxf(mx, gl[e * 16]);
        float s = 0.0f;
        #pragma unroll
        for (int e = 0; e < NEXP; ++e) s += __expf(gl[e * 16] - mx);
        float inv = __builtin_amdgcn_rcpf(s);
        #pragma unroll
        for (int e = 0; e < NEXP; ++e) gl[e * 16] = __expf(gl[e * 16] - mx) * inv;
    }
    __syncthreads();

    // ---- fused expert GEMMs + bias + GELU + gate-weighted accumulation ----
    // Per expert: hoist 8 B fragments (64 VGPRs) once, then retire the four
    // M-tiles sequentially (one 8-reg accumulator live at a time).
    // Both outer loops pinned rolled: unrolling them only doubles register
    // liveness (outAcc + in-flight bfrag) and pushes the wave past 256 VGPRs.
    #pragma unroll 1
    for (int half = 0; half < 2; ++half) {
        const int o = wave * 16 + half * 128;

        v8f outAcc[4];
        #pragma unroll
        for (int mt = 0; mt < 4; ++mt)
            #pragma unroll
            for (int r = 0; r < 8; ++r) outAcc[mt][r] = 0.0f;

        #pragma unroll 1
        for (int e = 0; e < NEXP; ++e) {
            const __bf16* bcol = Wt + ((size_t)e * ODIM + o + ln) * DDIM;

            if (e < NEXP - 1)                    // pull next expert's column toward L2/L0
                __builtin_prefetch(bcol + ODIM * DDIM, 0, 1);

            v16bf bfrag[8];
            #pragma unroll
            for (int ks = 0; ks < 8; ++ks) bfrag[ks] = load_frag_b(bcol, ks * 32, g);

            const float bias = be[e * ODIM + o + ln];

            #pragma unroll
            for (int mt = 0; mt < 4; ++mt) {
                const __bf16* arow = &xs[(mt * 16 + ln) * XS_STRIDE];

                v8f acc;
                #pragma unroll
                for (int r = 0; r < 8; ++r) acc[r] = 0.0f;

                #pragma unroll
                for (int ks = 0; ks < 8; ++ks) {
                    v16bf a = load_frag_a(arow, ks * 32, g);
                    acc = __builtin_amdgcn_wmma_f32_16x16x32_bf16(
                        false, a, false, bfrag[ks], (short)0, acc, false, false);
                }

                const float4* gp = (const float4*)(&gateLds[mt * 256 + e * 16 + 8 * g]);
                float4 gv0 = gp[0];
                float4 gv1 = gp[1];
                float gw[8] = { gv0.x, gv0.y, gv0.z, gv0.w, gv1.x, gv1.y, gv1.z, gv1.w };
                #pragma unroll
                for (int r = 0; r < 8; ++r)
                    outAcc[mt][r] = fmaf(gw[r], gelu_fast(acc[r] + bias), outAcc[mt][r]);
            }
        }

        // ---- store this half: token m = mt*16 + 8g + r, column o+ln ----
        #pragma unroll
        for (int mt = 0; mt < 4; ++mt)
            #pragma unroll
            for (int r = 0; r < 8; ++r) {
                const size_t row = (size_t)(tokBase + mt * 16 + 8 * g + r) * ODIM;
                out[row + o + ln] = outAcc[mt][r];
            }
    }
}

// ---------------- launcher ----------------

extern "C" void kernel_launch(void* const* d_in, const int* in_sizes, int n_in,
                              void* d_out, int out_size, void* d_ws, size_t ws_size,
                              hipStream_t stream) {
    const float* x  = (const float*)d_in[0];   // [4,4096,256]
    const float* We = (const float*)d_in[1];   // [16,256,256]
    const float* be = (const float*)d_in[2];   // [16,256]
    const float* Wg = (const float*)d_in[3];   // [256,16]
    const float* bg = (const float*)d_in[4];   // [16]
    float* out = (float*)d_out;                // [4,4096,256]

    __bf16* xb  = (__bf16*)d_ws;
    __bf16* Wt  = (__bf16*)((char*)d_ws + 8388608);
    __bf16* WgT = (__bf16*)((char*)d_ws + 10485760);

    moe_prep_x <<<NTOK * DDIM / 256, 256, 0, stream>>>(x, xb);
    moe_prep_we<<<NEXP * ODIM * DDIM / 256, 256, 0, stream>>>(We, Wt);
    moe_prep_wg<<<NEXP * DDIM / 256, 256, 0, stream>>>(Wg, WgT);

    moe_main<<<NTOK / 64, 256, 0, stream>>>(xb, Wt, WgT, be, bg, out);
}
